// DATwithMotionEstimation_68332929679390
// MI455X (gfx1250) — compile-verified, matrix operand order checked by reading, not run on hardware
//
#include <hip/hip_runtime.h>
#include <math.h>

typedef __attribute__((ext_vector_type(16))) _Float16 v16h;
typedef __attribute__((ext_vector_type(8)))  _Float16 v8h;
typedef __attribute__((ext_vector_type(8)))  float    v8f;
typedef __attribute__((ext_vector_type(4)))  int      v4i;

#define FHWc 4096
#define Cc   96
#define NHh  12
#define HIDc 192
#define SCALEc 0.35355339059327373f
#define KC   256

// ---------------- weight prep: f32 -> f16, transposed to [N][Kpad], zero-padded ----------------
// kmajor=1: src is [K][N] (token weights); kmajor=0: src is [N][K] (conv OIHW flattened)
__global__ void k_wprep(const float* __restrict__ w, int N, int K, int Kpad, int kmajor,
                        _Float16* __restrict__ out)
{
  long idx = (long)blockIdx.x * blockDim.x + threadIdx.x;
  if (idx >= (long)N * Kpad) return;
  int k = (int)(idx % Kpad), n = (int)(idx / Kpad);
  float v = 0.f;
  if (k < K) v = kmajor ? w[(size_t)k * N + n] : w[(size_t)n * K + k];
  out[idx] = (_Float16)v;
}

// ---------------- f16 im2col for 3x3 SAME conv ----------------
__global__ void k_im2col(const float* __restrict__ in, int CinTot, int ciOff, int Cin,
                         int Kpad, _Float16* __restrict__ out)
{
  long idx = (long)blockIdx.x * blockDim.x + threadIdx.x;
  if (idx >= (long)2 * FHWc * Kpad) return;
  int k = (int)(idx % Kpad);
  int m = (int)(idx / Kpad);
  int K = Cin * 9;
  float v = 0.f;
  if (k < K) {
    int b = m >> 12, f = m & 4095;
    int y = f >> 6, x = f & 63;
    int ci = k / 9, t = k - ci * 9;
    int dy = t / 3 - 1, dx = t - (t / 3) * 3 - 1;
    int yy = y + dy, xx = x + dx;
    if ((unsigned)yy < 64u && (unsigned)xx < 64u)
      v = in[(size_t)(b * CinTot + ciOff + ci) * FHWc + yy * 64 + xx];
  }
  out[idx] = (_Float16)v;
}

// ---------------- unified f16 WMMA GEMM, 16x32 output strip per wave ----------------
// C[M,N] = A[M,Kpad] @ BT[N,Kpad]^T (+bias)(prelu)(+resid); two N-tiles per wave share
// the A fragment (register reuse + back-to-back independent WMMAs for XDL pipelining);
// B tiles staged through LDS in KC-chunks via async-to-LDS when available.
__global__ __launch_bounds__(256)
void k_gemm16(const _Float16* __restrict__ A, int M, int Kpad,
              const _Float16* __restrict__ BT, int N,
              const float* __restrict__ bias, const float* __restrict__ pre_a,
              const float* __restrict__ residTok, const float* __restrict__ residCHW,
              float* __restrict__ C32, _Float16* __restrict__ C16,
              int outCHW, int OutTot, int coOff)
{
  __shared__ _Float16 lbT[32 * KC];   // 32 N-rows x KC K-columns, 16 KB
  const int lane = threadIdx.x & 31;
  const int wave = threadIdx.x >> 5;
  const int m0 = (blockIdx.x * 8 + wave) << 4;
  const int n0 = blockIdx.y << 5;     // 32 columns per block
  const int am = m0 + (lane & 15);
  const int bn = lane & 15;
  const int kb = (lane >> 4) << 3;

  v8f acc[2] = {};
  for (int kc = 0; kc < Kpad; kc += KC) {
    int klen = Kpad - kc;
    if (klen > KC) klen = KC;
    __syncthreads();
    {
      int units = klen >> 3;          // 8-half (16B) units per row
      int total = units << 5;         // 32 rows
      for (int u = threadIdx.x; u < total; u += 256) {
        int n = u / units, ko = (u - n * units) << 3;
        bool valid = (n0 + n) < N;
#if defined(__gfx1250__) && __has_builtin(__builtin_amdgcn_global_load_async_to_lds_b128)
        if (valid) {
          __builtin_amdgcn_global_load_async_to_lds_b128(
              (__attribute__((address_space(1))) v4i*)&BT[(size_t)(n0 + n) * Kpad + kc + ko],
              (__attribute__((address_space(3))) v4i*)&lbT[n * KC + ko], 0, 0);
        } else {
          v8h z = {};
          *(v8h*)&lbT[n * KC + ko] = z;
        }
#else
        v8h val = {};
        if (valid) val = *(const v8h*)&BT[(size_t)(n0 + n) * Kpad + kc + ko];
        *(v8h*)&lbT[n * KC + ko] = val;
#endif
      }
#if defined(__gfx1250__) && __has_builtin(__builtin_amdgcn_global_load_async_to_lds_b128) && __has_builtin(__builtin_amdgcn_s_wait_asynccnt)
      __builtin_amdgcn_s_wait_asynccnt(0);
#endif
    }
    __syncthreads();
    if (m0 < M) {
      for (int k0 = 0; k0 < klen; k0 += 32) {
        const _Float16* ap = &A[(size_t)am * Kpad + kc + k0 + kb];
        if (k0 + 64 < klen) __builtin_prefetch(ap + 64, 0, 1);
        v8h alo = *(const v8h*)ap;
        v8h ahi = *(const v8h*)(ap + 16);
        v16h a = __builtin_shufflevector(alo, ahi, 0, 1, 2, 3, 4, 5, 6, 7, 8, 9, 10, 11, 12, 13, 14, 15);
        // tile 0: LDS rows 0..15 ; tile 1: LDS rows 16..31 — same A fragment
        v8h b0lo = *(const v8h*)&lbT[bn * KC + k0 + kb];
        v8h b0hi = *(const v8h*)&lbT[bn * KC + k0 + 16 + kb];
        v8h b1lo = *(const v8h*)&lbT[(bn + 16) * KC + k0 + kb];
        v8h b1hi = *(const v8h*)&lbT[(bn + 16) * KC + k0 + 16 + kb];
        v16h b0 = __builtin_shufflevector(b0lo, b0hi, 0, 1, 2, 3, 4, 5, 6, 7, 8, 9, 10, 11, 12, 13, 14, 15);
        v16h b1 = __builtin_shufflevector(b1lo, b1hi, 0, 1, 2, 3, 4, 5, 6, 7, 8, 9, 10, 11, 12, 13, 14, 15);
        acc[0] = __builtin_amdgcn_wmma_f32_16x16x32_f16(false, a, false, b0, (short)0, acc[0], false, false);
        acc[1] = __builtin_amdgcn_wmma_f32_16x16x32_f16(false, a, false, b1, (short)0, acc[1], false, false);
      }
    }
  }
  if (m0 >= M) return;
#pragma unroll
  for (int t = 0; t < 2; ++t) {
    const int gn = n0 + t * 16 + bn;
    if (gn >= N) continue;
    const float bi = bias ? bias[gn] : 0.f;
    const float pa = pre_a ? pre_a[gn] : 1.f;   // slope 1 == identity
#pragma unroll
    for (int v = 0; v < 8; ++v) {
      int m = m0 + v + ((lane >> 4) << 3);
      int b = m >> 12, f = m & 4095;
      float r = acc[t][v] + bi;
      r = r >= 0.f ? r : pa * r;
      if (residTok) r += residTok[(size_t)m * N + gn];
      if (residCHW) r += residCHW[(size_t)(b * OutTot + coOff + gn) * FHWc + f];
      if (outCHW) C32[(size_t)(b * OutTot + coOff + gn) * FHWc + f] = r;
      else if (C16) C16[(size_t)m * N + gn] = (_Float16)r;
      else C32[(size_t)m * N + gn] = r;
    }
  }
}

// ---------------- warp + concat [feat_t, bwarp(featx, ftx), ftx] ----------------
__global__ void k_build_cat(const float* __restrict__ ft_t, const float* __restrict__ fx,
                            const float* __restrict__ flo, float* __restrict__ cat)
{
  int idx = blockIdx.x * blockDim.x + threadIdx.x;
  if (idx >= 2 * FHWc) return;
  int b = idx >> 12, f = idx & 4095;
  int x = f & 63, y = f >> 6;
  float u = flo[(size_t)(b * 2 + 0) * FHWc + f];
  float v = flo[(size_t)(b * 2 + 1) * FHWc + f];
  float px = x + u, py = y + v;
  float fx0 = floorf(px), fy0 = floorf(py);
  float wx = px - fx0, wy = py - fy0;
  int x0 = (int)fx0, y0 = (int)fy0;
  float ww[4] = {(1 - wx) * (1 - wy), wx * (1 - wy), (1 - wx) * wy, wx * wy};
  int cx[4] = {x0, x0 + 1, x0, x0 + 1};
  int cy[4] = {y0, y0, y0 + 1, y0 + 1};
  float cw[4]; int cidx[4];
#pragma unroll
  for (int q = 0; q < 4; ++q) {
    bool ok = (unsigned)cx[q] < 64u && (unsigned)cy[q] < 64u;
    cw[q] = ok ? ww[q] : 0.f;
    cidx[q] = ok ? (cy[q] * 64 + cx[q]) : 0;
  }
  for (int c = 0; c < Cc; ++c) {
    cat[(size_t)(b * 194 + c) * FHWc + f] = ft_t[(size_t)(b * Cc + c) * FHWc + f];
    const float* base = fx + (size_t)(b * Cc + c) * FHWc;
    float s = cw[0] * base[cidx[0]] + cw[1] * base[cidx[1]] +
              cw[2] * base[cidx[2]] + cw[3] * base[cidx[3]];
    cat[(size_t)(b * 194 + 96 + c) * FHWc + f] = s;
  }
  cat[(size_t)(b * 194 + 192) * FHWc + f] = u;
  cat[(size_t)(b * 194 + 193) * FHWc + f] = v;
}

__global__ void k_copy48(const float* __restrict__ src, float* __restrict__ dst)
{
  int idx = blockIdx.x * blockDim.x + threadIdx.x;
  if (idx >= 2 * 48 * FHWc) return;
  int f = idx & 4095, r = idx >> 12;
  int c = r % 48, b = r / 48;
  size_t i = (size_t)(b * Cc + c) * FHWc + f;
  dst[i] = src[i];
}

__global__ void k_flow(const float* __restrict__ off, const float* __restrict__ flo,
                       float* __restrict__ flow)
{
  int idx = blockIdx.x * blockDim.x + threadIdx.x;
  if (idx >= 2 * 108 * 2 * FHWc) return;
  int f = idx & 4095, t = idx >> 12;
  int d = t & 1, bg = t >> 1;
  int b = bg / 108, gs = bg - b * 108;
  float o = off[(size_t)(b * 216 + gs * 2 + d) * FHWc + f];
  flow[idx] = 2.f * tanhf(o) + flo[(size_t)(b * 2 + d) * FHWc + f];
}

// LN over channels; CHW out (tokens=0) or token-major out32/out16 (tokens=1)
__global__ void k_layernorm_chw(const float* __restrict__ in, const float* __restrict__ g,
                                const float* __restrict__ be, float* __restrict__ out32,
                                _Float16* __restrict__ out16, int tokens)
{
  int idx = blockIdx.x * blockDim.x + threadIdx.x;
  if (idx >= 2 * FHWc) return;
  int b = idx >> 12, f = idx & 4095;
  float mu = 0.f;
  for (int c = 0; c < Cc; ++c) mu += in[(size_t)(b * Cc + c) * FHWc + f];
  mu *= (1.f / Cc);
  float var = 0.f;
  for (int c = 0; c < Cc; ++c) {
    float d = in[(size_t)(b * Cc + c) * FHWc + f] - mu;
    var += d * d;
  }
  var *= (1.f / Cc);
  float rs = rsqrtf(var + 1e-5f);
  for (int c = 0; c < Cc; ++c) {
    float r = (in[(size_t)(b * Cc + c) * FHWc + f] - mu) * rs * g[c] + be[c];
    if (tokens) {
      out32[(size_t)idx * Cc + c] = r;
      if (out16) out16[(size_t)idx * Cc + c] = (_Float16)r;
    } else {
      out32[(size_t)(b * Cc + c) * FHWc + f] = r;
    }
  }
}

// deformable bilinear sampling -> kv tokens (f16) + coords
__global__ void k_sample(const float* __restrict__ fn, const float* __restrict__ flow,
                         _Float16* __restrict__ kvt, float* __restrict__ coords, int xsel)
{
  int idx = blockIdx.x * blockDim.x + threadIdx.x;
  if (idx >= 2 * 108 * FHWc) return;
  int f = idx & 4095, r = idx >> 12;
  int gs = r % 108, b = r / 108;
  int g = gs / 9, s = gs - g * 9;
  int x = f & 63, y = f >> 6;
  float fxv = flow[(size_t)((b * 108 + gs) * 2 + 0) * FHWc + f];
  float fyv = flow[(size_t)((b * 108 + gs) * 2 + 1) * FHWc + f];
  float gx = (2.f * x / 63.f - 1.f) + fxv / 126.f;   // /(FW-1)/2
  float gy = (2.f * y / 63.f - 1.f) + fyv / 31.5f;   // /((FH-1)/2)
  coords[((size_t)(b * 108 + gs) * FHWc + f) * 2 + 0] = gx;
  coords[((size_t)(b * 108 + gs) * FHWc + f) * 2 + 1] = gy;
  float px = (gx + 1.f) * 0.5f * 63.f;
  float py = (gy + 1.f) * 0.5f * 63.f;
  float fx0 = floorf(px), fy0 = floorf(py);
  float wx = px - fx0, wy = py - fy0;
  int x0 = (int)fx0, y0 = (int)fy0;
  float ww[4] = {(1 - wx) * (1 - wy), wx * (1 - wy), (1 - wx) * wy, wx * wy};
  int cx[4] = {x0, x0 + 1, x0, x0 + 1};
  int cy[4] = {y0, y0, y0 + 1, y0 + 1};
  float cw[4]; int cidx[4];
#pragma unroll
  for (int q = 0; q < 4; ++q) {
    bool ok = (unsigned)cx[q] < 64u && (unsigned)cy[q] < 64u;
    cw[q] = ok ? ww[q] : 0.f;
    cidx[q] = ok ? (cy[q] * 64 + cx[q]) : 0;
  }
  size_t tok = ((size_t)(b * 18 + xsel * 9 + s) * FHWc + f) * Cc;
#pragma unroll
  for (int cc = 0; cc < 8; ++cc) {
    int c = g * 8 + cc;
    const float* base = fn + (size_t)(b * Cc + c) * FHWc;
    float v = cw[0] * base[cidx[0]] + cw[1] * base[cidx[1]] +
              cw[2] * base[cidx[2]] + cw[3] * base[cidx[3]];
    kvt[tok + c] = (_Float16)v;
  }
}

// fused attention: logits, softmax(18), softmax(9|9) scores, attn_out + qn residual
__global__ void k_attn(const float* __restrict__ qp, const _Float16* __restrict__ kp,
                       const _Float16* __restrict__ vp, const float* __restrict__ qn,
                       float* __restrict__ tok, float* __restrict__ sc0, float* __restrict__ sc1)
{
  int idx = blockIdx.x * blockDim.x + threadIdx.x;
  if (idx >= 2 * FHWc * NHh) return;
  int h = idx % NHh, m = idx / NHh;
  int b = m >> 12, f = m & 4095;
  float q[8];
#pragma unroll
  for (int c = 0; c < 8; ++c) q[c] = qp[(size_t)m * Cc + h * 8 + c];
  float l[18];
  for (int s = 0; s < 18; ++s) {
    size_t base = ((size_t)(b * 18 + s) * FHWc + f) * Cc + h * 8;
    float d = 0.f;
#pragma unroll
    for (int c = 0; c < 8; ++c) d += q[c] * (float)kp[base + c];
    l[s] = d * SCALEc;
  }
  float mx = l[0];
  for (int s = 1; s < 18; ++s) mx = fmaxf(mx, l[s]);
  float sum = 0.f, sc[18];
  for (int s = 0; s < 18; ++s) { sc[s] = __expf(l[s] - mx); sum += sc[s]; }
  float inv = 1.f / sum;
  float acc[8] = {0, 0, 0, 0, 0, 0, 0, 0};
  for (int s = 0; s < 18; ++s) {
    float w = sc[s] * inv;
    size_t base = ((size_t)(b * 18 + s) * FHWc + f) * Cc + h * 8;
#pragma unroll
    for (int c = 0; c < 8; ++c) acc[c] += w * (float)vp[base + c];
  }
#pragma unroll
  for (int c = 0; c < 8; ++c)
    tok[(size_t)m * Cc + h * 8 + c] = qn[(size_t)m * Cc + h * 8 + c] + acc[c];
  float m0 = l[0]; for (int s = 1; s < 9; ++s) m0 = fmaxf(m0, l[s]);
  float s0 = 0.f, e0[9];
  for (int s = 0; s < 9; ++s) { e0[s] = __expf(l[s] - m0); s0 += e0[s]; }
  float i0 = 1.f / s0;
  float m1 = l[9]; for (int s = 10; s < 18; ++s) m1 = fmaxf(m1, l[s]);
  float s1 = 0.f, e1[9];
  for (int s = 0; s < 9; ++s) { e1[s] = __expf(l[9 + s] - m1); s1 += e1[s]; }
  float i1 = 1.f / s1;
  for (int s = 0; s < 9; ++s) {
    sc0[(size_t)(b * 108 + h * 9 + s) * FHWc + f] = e0[s] * i0;
    sc1[(size_t)(b * 108 + h * 9 + s) * FHWc + f] = e1[s] * i1;
  }
}

__global__ void k_ln_tokens(const float* __restrict__ in, const float* __restrict__ g,
                            const float* __restrict__ be, _Float16* __restrict__ out16)
{
  int m = blockIdx.x * blockDim.x + threadIdx.x;
  if (m >= 2 * FHWc) return;
  float mu = 0.f;
  for (int c = 0; c < Cc; ++c) mu += in[(size_t)m * Cc + c];
  mu *= (1.f / Cc);
  float var = 0.f;
  for (int c = 0; c < Cc; ++c) { float d = in[(size_t)m * Cc + c] - mu; var += d * d; }
  var *= (1.f / Cc);
  float rs = rsqrtf(var + 1e-5f);
  for (int c = 0; c < Cc; ++c)
    out16[(size_t)m * Cc + c] = (_Float16)((in[(size_t)m * Cc + c] - mu) * rs * g[c] + be[c]);
}

__global__ void k_dwconv_gelu(const float* __restrict__ h1, const float* __restrict__ w,
                              const float* __restrict__ bias, _Float16* __restrict__ out16)
{
  int idx = blockIdx.x * blockDim.x + threadIdx.x;
  if (idx >= 2 * FHWc * HIDc) return;
  int c = idx % HIDc, m = idx / HIDc;
  int b = m >> 12, f = m & 4095;
  int x = f & 63, y = f >> 6;
  float acc = bias[c];
#pragma unroll
  for (int t = 0; t < 9; ++t) {
    int dy = t / 3 - 1, dx = t - (t / 3) * 3 - 1;
    int yy = y + dy, xx = x + dx;
    if ((unsigned)yy < 64u && (unsigned)xx < 64u)
      acc += w[c * 9 + t] * h1[((size_t)(b * FHWc + yy * 64 + xx)) * HIDc + c];
  }
  out16[(size_t)m * HIDc + c] = (_Float16)(0.5f * acc * (1.f + erff(acc * 0.70710678118654752f)));
}

__global__ void k_moved(const float* __restrict__ sc, const float* __restrict__ coords,
                        float* __restrict__ moved)
{
  int idx = blockIdx.x * blockDim.x + threadIdx.x;
  if (idx >= 2 * FHWc) return;
  int b = idx >> 12, f = idx & 4095;
  float mx = 0.f, my = 0.f;
  for (int s = 0; s < 108; ++s) {
    float w = sc[(size_t)(b * 108 + s) * FHWc + f];
    size_t ci = ((size_t)(b * 108 + s) * FHWc + f) * 2;
    mx += w * coords[ci];
    my += w * coords[ci + 1];
  }
  moved[(size_t)idx * 2 + 0] = mx;
  moved[(size_t)idx * 2 + 1] = my;
}

// (moved - base_p) @ mot_w + mot_b  ==  (mgx-x)*A0 + (mgy-y)*A1 + Kc
__global__ void k_mprec(const float* __restrict__ cw, const float* __restrict__ cb,
                        const float* __restrict__ mw, const float* __restrict__ mb,
                        float* __restrict__ prec)
{
  int c2 = blockIdx.x * blockDim.x + threadIdx.x;
  if (c2 >= Cc) return;
  float a0 = 0.f, a1 = 0.f, kc = 0.f;
  for (int c = 0; c < Cc; ++c) {
    float m = mw[(size_t)c * Cc + c2];
    a0 += cw[c] * m;
    a1 += cw[Cc + c] * m;
    kc += cb[c] * m;
  }
  prec[c2] = a0;
  prec[Cc + c2] = a1;
  prec[2 * Cc + c2] = 11.f * kc + mb[c2];
}

__global__ void k_motion(const float* __restrict__ moved, const float* __restrict__ prec,
                         float* __restrict__ out)
{
  int idx = blockIdx.x * blockDim.x + threadIdx.x;
  if (idx >= 2 * FHWc * Cc) return;
  int c2 = idx % Cc, m = idx / Cc;
  int f = m & 4095;
  float dx = moved[(size_t)m * 2 + 0] - (float)(f & 63);
  float dy = moved[(size_t)m * 2 + 1] - (float)(f >> 6);
  out[(size_t)m * Cc + c2] = dx * prec[c2] + dy * prec[Cc + c2] + prec[2 * Cc + c2];
}

// ------------------------------- launch -------------------------------
extern "C" void kernel_launch(void* const* d_in, const int* in_sizes, int n_in,
                              void* d_out, int out_size, void* d_ws, size_t ws_size,
                              hipStream_t stream) {
  const float* feat_t = (const float*)d_in[0];
  const float* feat0  = (const float*)d_in[1];
  const float* feat1  = (const float*)d_in[2];
  const float* ft0    = (const float*)d_in[3];
  const float* ft1    = (const float*)d_in[4];
  const float* n1g = (const float*)d_in[5];
  const float* n1b = (const float*)d_in[6];
  const float* n2g = (const float*)d_in[7];
  const float* n2b = (const float*)d_in[8];
  const float* q_w = (const float*)d_in[9];
  const float* q_b = (const float*)d_in[10];
  const float* k_w = (const float*)d_in[11];
  const float* k_b = (const float*)d_in[12];
  const float* v_w = (const float*)d_in[13];
  const float* v_b = (const float*)d_in[14];
  const float* crf_w = (const float*)d_in[15];
  const float* crf_b = (const float*)d_in[16];
  const float* crf_a = (const float*)d_in[17];
  const float* rb1_w = (const float*)d_in[18];
  const float* rb1_b = (const float*)d_in[19];
  const float* rb1_a = (const float*)d_in[20];
  const float* rb2_w = (const float*)d_in[21];
  const float* rb2_b = (const float*)d_in[22];
  const float* rb2_a = (const float*)d_in[23];
  const float* rb3_w = (const float*)d_in[24];
  const float* rb3_b = (const float*)d_in[25];
  const float* rb3_a = (const float*)d_in[26];
  const float* off_w = (const float*)d_in[27];
  const float* off_b = (const float*)d_in[28];
  const float* fc1_w = (const float*)d_in[29];
  const float* fc1_b = (const float*)d_in[30];
  const float* dw_w  = (const float*)d_in[31];
  const float* dw_b  = (const float*)d_in[32];
  const float* fc2_w = (const float*)d_in[33];
  const float* fc2_b = (const float*)d_in[34];
  const float* coord_w = (const float*)d_in[35];
  const float* coord_b = (const float*)d_in[36];
  const float* mot_w = (const float*)d_in[37];
  const float* mot_b = (const float*)d_in[38];

  float* out = (float*)d_out;
  const float* FNUL = nullptr;
  _Float16* HNUL = nullptr;

  char* ws = (char*)d_ws;
  size_t pos = 0;
  auto alloc = [&](size_t bytes) -> char* {
    char* p = ws + pos;
    pos += (bytes + 255) & ~(size_t)255;
    return p;
  };
  const size_t F32MAP = (size_t)2 * 96 * 4096 * 4;
  // f32 intermediates
  float*    cat   = (float*)alloc((size_t)2 * 194 * 4096 * 4);
  float*    xb    = (float*)alloc(F32MAP);
  float*    ob    = (float*)alloc(F32MAP);
  float*    o2    = (float*)alloc(F32MAP);
  float*    mv    = (float*)alloc(F32MAP);
  float*    offb  = (float*)alloc((size_t)2 * 216 * 4096 * 4);
  float*    flowb = (float*)alloc((size_t)2 * 108 * 2 * 4096 * 4);
  float*    fnorm = (float*)alloc(F32MAP);
  float*    qn32  = (float*)alloc(F32MAP);
  float*    coords0 = (float*)alloc((size_t)2 * 108 * 4096 * 2 * 4);
  float*    coords1 = (float*)alloc((size_t)2 * 108 * 4096 * 2 * 4);
  float*    qp    = (float*)alloc(F32MAP);
  float*    sc0   = (float*)alloc((size_t)2 * 108 * 4096 * 4);
  float*    sc1   = (float*)alloc((size_t)2 * 108 * 4096 * 4);
  float*    tok   = (float*)alloc(F32MAP);
  float*    h1    = (float*)alloc((size_t)2 * 4096 * 192 * 4);
  float*    moved0 = (float*)alloc((size_t)2 * 4096 * 2 * 4);
  float*    moved1 = (float*)alloc((size_t)2 * 4096 * 2 * 4);
  float*    prec  = (float*)alloc((size_t)3 * 96 * 4);
  // f16 intermediates
  _Float16* kvt   = (_Float16*)alloc((size_t)2 * 18 * 4096 * 96 * 2);
  _Float16* kp    = (_Float16*)alloc((size_t)2 * 18 * 4096 * 96 * 2);
  _Float16* vp    = (_Float16*)alloc((size_t)2 * 18 * 4096 * 96 * 2);
  _Float16* qn16  = (_Float16*)alloc((size_t)2 * 4096 * 96 * 2);
  _Float16* ln216 = (_Float16*)alloc((size_t)2 * 4096 * 96 * 2);
  _Float16* hact16 = (_Float16*)alloc((size_t)2 * 4096 * 192 * 2);
  _Float16* im2c  = (_Float16*)alloc((size_t)2 * 4096 * 1760 * 2);
  // prepped f16 transposed weights [N][Kpad]
  _Float16* crfT = (_Float16*)alloc((size_t)96 * 1760 * 2);
  _Float16* rb1T = (_Float16*)alloc((size_t)96 * 864 * 2);
  _Float16* rb2T = (_Float16*)alloc((size_t)48 * 448 * 2);
  _Float16* rb3T = (_Float16*)alloc((size_t)96 * 864 * 2);
  _Float16* offT = (_Float16*)alloc((size_t)216 * 864 * 2);
  _Float16* qT   = (_Float16*)alloc((size_t)96 * 96 * 2);
  _Float16* kT   = (_Float16*)alloc((size_t)96 * 96 * 2);
  _Float16* vT   = (_Float16*)alloc((size_t)96 * 96 * 2);
  _Float16* fc1T = (_Float16*)alloc((size_t)192 * 96 * 2);
  _Float16* fc2T = (_Float16*)alloc((size_t)96 * 192 * 2);

  const dim3 blk(256);
  auto grid1 = [](long n) { return dim3((unsigned)((n + 255) / 256)); };

  // ---- weight prep (once) ----
  hipLaunchKernelGGL(k_wprep, grid1((long)96 * 1760), blk, 0, stream, crf_w, 96, 1746, 1760, 0, crfT);
  hipLaunchKernelGGL(k_wprep, grid1((long)96 * 864), blk, 0, stream, rb1_w, 96, 864, 864, 0, rb1T);
  hipLaunchKernelGGL(k_wprep, grid1((long)48 * 448), blk, 0, stream, rb2_w, 48, 432, 448, 0, rb2T);
  hipLaunchKernelGGL(k_wprep, grid1((long)96 * 864), blk, 0, stream, rb3_w, 96, 864, 864, 0, rb3T);
  hipLaunchKernelGGL(k_wprep, grid1((long)216 * 864), blk, 0, stream, off_w, 216, 864, 864, 0, offT);
  hipLaunchKernelGGL(k_wprep, grid1((long)96 * 96), blk, 0, stream, q_w, 96, 96, 96, 1, qT);
  hipLaunchKernelGGL(k_wprep, grid1((long)96 * 96), blk, 0, stream, k_w, 96, 96, 96, 1, kT);
  hipLaunchKernelGGL(k_wprep, grid1((long)96 * 96), blk, 0, stream, v_w, 96, 96, 96, 1, vT);
  hipLaunchKernelGGL(k_wprep, grid1((long)192 * 96), blk, 0, stream, fc1_w, 192, 96, 96, 1, fc1T);
  hipLaunchKernelGGL(k_wprep, grid1((long)96 * 192), blk, 0, stream, fc2_w, 96, 192, 192, 1, fc2T);

  // ---- per-source movement + offsets + sampling ----
  for (int xs = 0; xs < 2; ++xs) {
    const float* fx = xs ? feat1 : feat0;
    const float* fl = xs ? ft1 : ft0;
    float* coords = xs ? coords1 : coords0;

    hipLaunchKernelGGL(k_build_cat, grid1(2 * 4096), blk, 0, stream, feat_t, fx, fl, cat);
    hipLaunchKernelGGL(k_im2col, grid1((long)2 * 4096 * 1760), blk, 0, stream, cat, 194, 0, 194, 1760, im2c);
    hipLaunchKernelGGL(k_gemm16, dim3(64, 3), blk, 0, stream, im2c, 2 * 4096, 1760, crfT, 96,
                       crf_b, crf_a, FNUL, FNUL, xb, HNUL, 1, 96, 0);
    hipLaunchKernelGGL(k_im2col, grid1((long)2 * 4096 * 864), blk, 0, stream, xb, 96, 0, 96, 864, im2c);
    hipLaunchKernelGGL(k_gemm16, dim3(64, 3), blk, 0, stream, im2c, 2 * 4096, 864, rb1T, 96,
                       rb1_b, rb1_a, FNUL, FNUL, ob, HNUL, 1, 96, 0);
    hipLaunchKernelGGL(k_copy48, grid1((long)2 * 48 * 4096), blk, 0, stream, ob, o2);
    hipLaunchKernelGGL(k_im2col, grid1((long)2 * 4096 * 448), blk, 0, stream, ob, 96, 48, 48, 448, im2c);
    hipLaunchKernelGGL(k_gemm16, dim3(64, 2), blk, 0, stream, im2c, 2 * 4096, 448, rb2T, 48,
                       rb2_b, rb2_a, FNUL, FNUL, o2, HNUL, 1, 96, 48);
    hipLaunchKernelGGL(k_im2col, grid1((long)2 * 4096 * 864), blk, 0, stream, o2, 96, 0, 96, 864, im2c);
    hipLaunchKernelGGL(k_gemm16, dim3(64, 3), blk, 0, stream, im2c, 2 * 4096, 864, rb3T, 96,
                       rb3_b, rb3_a, FNUL, xb, mv, HNUL, 1, 96, 0);
    hipLaunchKernelGGL(k_im2col, grid1((long)2 * 4096 * 864), blk, 0, stream, mv, 96, 0, 96, 864, im2c);
    hipLaunchKernelGGL(k_gemm16, dim3(64, 7), blk, 0, stream, im2c, 2 * 4096, 864, offT, 216,
                       off_b, FNUL, FNUL, FNUL, offb, HNUL, 1, 216, 0);
    hipLaunchKernelGGL(k_flow, grid1((long)2 * 108 * 2 * 4096), blk, 0, stream, offb, fl, flowb);
    hipLaunchKernelGGL(k_layernorm_chw, grid1(2 * 4096), blk, 0, stream, fx, n1g, n1b, fnorm, HNUL, 0);
    hipLaunchKernelGGL(k_sample, grid1((long)2 * 108 * 4096), blk, 0, stream, fnorm, flowb, kvt, coords, xs);
  }

  // ---- attention ----
  hipLaunchKernelGGL(k_layernorm_chw, grid1(2 * 4096), blk, 0, stream, feat_t, n1g, n1b, qn32, qn16, 1);
  hipLaunchKernelGGL(k_gemm16, dim3(64, 3), blk, 0, stream, qn16, 2 * 4096, 96, qT, 96,
                     q_b, FNUL, FNUL, FNUL, qp, HNUL, 0, 0, 0);
  hipLaunchKernelGGL(k_gemm16, dim3(1152, 3), blk, 0, stream, kvt, 2 * 18 * 4096, 96, kT, 96,
                     k_b, FNUL, FNUL, FNUL, (float*)nullptr, kp, 0, 0, 0);
  hipLaunchKernelGGL(k_gemm16, dim3(1152, 3), blk, 0, stream, kvt, 2 * 18 * 4096, 96, vT, 96,
                     v_b, FNUL, FNUL, FNUL, (float*)nullptr, vp, 0, 0, 0);
  hipLaunchKernelGGL(k_attn, grid1((long)2 * 4096 * 12), blk, 0, stream, qp, kp, vp, qn32, tok, sc0, sc1);

  // ---- FFN ----
  hipLaunchKernelGGL(k_ln_tokens, grid1(2 * 4096), blk, 0, stream, tok, n2g, n2b, ln216);
  hipLaunchKernelGGL(k_gemm16, dim3(64, 6), blk, 0, stream, ln216, 2 * 4096, 96, fc1T, 192,
                     fc1_b, FNUL, FNUL, FNUL, h1, HNUL, 0, 0, 0);
  hipLaunchKernelGGL(k_dwconv_gelu, grid1((long)2 * 4096 * 192), blk, 0, stream, h1, dw_w, dw_b, hact16);
  hipLaunchKernelGGL(k_gemm16, dim3(64, 3), blk, 0, stream, hact16, 2 * 4096, 192, fc2T, 96,
                     fc2_b, FNUL, tok, FNUL, out, HNUL, 1, 96, 0);

  // ---- motion heads ----
  hipLaunchKernelGGL(k_moved, grid1(2 * 4096), blk, 0, stream, sc0, coords0, moved0);
  hipLaunchKernelGGL(k_moved, grid1(2 * 4096), blk, 0, stream, sc1, coords1, moved1);
  hipLaunchKernelGGL(k_mprec, dim3(1), dim3(128), 0, stream, coord_w, coord_b, mot_w, mot_b, prec);
  hipLaunchKernelGGL(k_motion, grid1((long)2 * 4096 * 96), blk, 0, stream,
                     moved0, prec, out + (size_t)2 * 96 * 4096);
  hipLaunchKernelGGL(k_motion, grid1((long)2 * 4096 * 96), blk, 0, stream,
                     moved1, prec, out + (size_t)2 * 2 * 96 * 4096);
}
